// PGExplainer_7275674599749
// MI455X (gfx1250) — compile-verified, matrix operand order checked by reading, not run on hardware
//
#include <hip/hip_runtime.h>
#include <hip/hip_bf16.h>
#include <math.h>

typedef __attribute__((ext_vector_type(16))) _Float16 v16h;
typedef __attribute__((ext_vector_type(8)))  _Float16 v8h;
typedef __attribute__((ext_vector_type(8)))  float    v8f;

#define F_DIM 128
#define H_DIM 64
#define TEMP_INV 0.2f
#define EPS_G 1e-8f

// ---------------------------------------------------------------------------
// Prep 1: f32 -> f16 embedding conversion (bandwidth trivial: ~77 MB total)
// ---------------------------------------------------------------------------
__global__ void pgx_cvt_f16(const float* __restrict__ src,
                            _Float16* __restrict__ dst, long long count) {
    long long i = (long long)blockIdx.x * blockDim.x + threadIdx.x;
    long long stride = (long long)gridDim.x * blockDim.x;
    for (; i < count; i += stride) dst[i] = (_Float16)src[i];
}

// ---------------------------------------------------------------------------
// Prep 2: swizzle W [K_total x 64] (row-major f32) into per-lane WMMA
// B-fragment layout, f16.  Fragment (nt, c) for wave32:
//   lane l: column n = nt*16 + (l&15); kbase = c*32 + (l>>4)*8
//   halves j=0..7  -> W[kbase+j][n]        (VGPR 0..3)
//   halves j=8..15 -> W[kbase+16+(j-8)][n] (VGPR 4..7)
// ---------------------------------------------------------------------------
__global__ void pgx_pack_w(const float* __restrict__ W,
                           _Float16* __restrict__ P, int nchunks) {
    int idx = blockIdx.x * blockDim.x + threadIdx.x;
    int total = 4 * nchunks * 32 * 16;
    if (idx >= total) return;
    int j    = idx & 15;
    int lane = (idx >> 4) & 31;
    int fc   = idx >> 9;            // nt*nchunks + c
    int c    = fc % nchunks;
    int nt   = fc / nchunks;
    int kbase = c * 32 + ((lane >> 4) & 1) * 8;
    int k = (j < 8) ? (kbase + j) : (kbase + 16 + (j - 8));
    int n = nt * 16 + (lane & 15);
    P[idx] = (_Float16)W[k * H_DIM + n];
}

// ---------------------------------------------------------------------------
// Main: 8 waves/block, each wave computes a 16-edge tile with WMMA.
// Edge-index loads are software-pipelined one tile ahead.
// ---------------------------------------------------------------------------
__global__ __launch_bounds__(256) void pgx_main(
    const _Float16* __restrict__ embh,   // [N][128] f16
    const int*      __restrict__ ei,     // [2*E]
    const float*    __restrict__ u,      // [E]
    const _Float16* __restrict__ w1p,    // packed 4*8*32*16 halves
    const _Float16* __restrict__ w2p,    // packed 4*2*32*16 halves
    const float*    __restrict__ b1,     // [64]
    const float*    __restrict__ b2,     // [64]
    const float*    __restrict__ w3,     // [64]
    const float*    __restrict__ b3,     // [1]
    float*          __restrict__ out,    // [E]
    int E, int numTiles)
{
    __shared__ _Float16 sW1[4 * 8 * 32 * 16];   // 32 KB
    __shared__ _Float16 sW2[4 * 2 * 32 * 16];   //  8 KB
    __shared__ _Float16 sH1[8][16 * H_DIM];     // 16 KB: per-wave staging

    const int tid = threadIdx.x;
    // cooperative weight staging into LDS (b128 copies)
    {
        const uint4* s1 = (const uint4*)w1p;  uint4* d1 = (uint4*)sW1;
        const uint4* s2 = (const uint4*)w2p;  uint4* d2 = (uint4*)sW2;
        for (int i = tid; i < (4*8*32*16) / 8; i += 256) d1[i] = s1[i];
        for (int i = tid; i < (4*2*32*16) / 8; i += 256) d2[i] = s2[i];
    }
    __syncthreads();

    const int wave = tid >> 5;
    const int lane = tid & 31;
    const int m    = lane & 15;       // edge row within tile / column id
    const int hi   = lane >> 4;       // half-wave select
    const int koff = hi * 8;          // K sub-offset per ISA A/B layout

    float rb1[4], rb2[4], rw3[4];
#pragma unroll
    for (int nt = 0; nt < 4; ++nt) {
        rb1[nt] = b1[nt * 16 + m];
        rb2[nt] = b2[nt * 16 + m];
        rw3[nt] = w3[nt * 16 + m];
    }
    const float bias3 = b3[0];

    _Float16* stage = &sH1[wave][0];

    const int tstride = gridDim.x * 8;
    int t = blockIdx.x * 8 + wave;

    // Prologue: preload indices for the first tile (branchless clamp).
    int nrow, ncol;
    {
        long long e0 = (long long)t * 16 + m;
        if (e0 >= E) e0 = (long long)E - 1;
        nrow = ei[e0];
        ncol = ei[(long long)E + e0];
    }

    for (; t < numTiles; t += tstride) {
        int e = t * 16 + m;
        if (e >= E) e = E - 1;                  // clamp tail (wave-internal)
        const float uval = u[e];                // coalesced, overlaps compute
        const _Float16* rowp = embh + (long long)nrow * F_DIM;
        const _Float16* colp = embh + (long long)ncol * F_DIM;

        // Pipeline: issue next tile's index loads now; consumed next iter.
        int nrow_n, ncol_n;
        {
            long long en = (long long)(t + tstride) * 16 + m;
            if (en >= E) en = (long long)E - 1;
            nrow_n = ei[en];
            ncol_n = ei[(long long)E + en];
        }

        // ---------------- layer 1: [16x256] x [256x64] -------------------
        v8f acc[4];
#pragma unroll
        for (int nt = 0; nt < 4; ++nt) acc[nt] = (v8f)(0.0f);

#pragma unroll
        for (int c = 0; c < 8; ++c) {
            const _Float16* base = ((c < 4) ? rowp : colp) + ((c & 3) * 32 + koff);
            union { v16h v; v8h h[2]; } A;
            A.h[0] = *(const v8h*)(base);        // K = kbase .. kbase+7
            A.h[1] = *(const v8h*)(base + 16);   // K = kbase+16 .. kbase+23
#pragma unroll
            for (int nt = 0; nt < 4; ++nt) {
                const v16h B = *(const v16h*)&sW1[((nt * 8 + c) * 32 + lane) * 16];
                acc[nt] = __builtin_amdgcn_wmma_f32_16x16x32_f16(
                    false, A.v, false, B, (short)0, acc[nt], false, false);
            }
        }

        // bias + ReLU, stage h1 as f16 [16 rows][64 cols] in wave-private LDS
        asm volatile("s_wait_dscnt 0" ::: "memory");  // WAR vs prev-iter loads
#pragma unroll
        for (int nt = 0; nt < 4; ++nt) {
#pragma unroll
            for (int r = 0; r < 8; ++r) {
                float v = acc[nt][r] + rb1[nt];
                v = v > 0.0f ? v : 0.0f;
                const int rowM = hi ? (r + 8) : r;
                stage[rowM * H_DIM + nt * 16 + m] = (_Float16)v;
            }
        }
        asm volatile("s_wait_dscnt 0" ::: "memory");  // RAW: stores -> loads

        // ---------------- layer 2: [16x64] x [64x64] ---------------------
        v8f acc2[4];
#pragma unroll
        for (int nt = 0; nt < 4; ++nt) acc2[nt] = (v8f)(0.0f);

#pragma unroll
        for (int c = 0; c < 2; ++c) {
            const _Float16* sb = stage + m * H_DIM + c * 32 + koff;
            union { v16h v; v8h h[2]; } A;
            A.h[0] = *(const v8h*)(sb);
            A.h[1] = *(const v8h*)(sb + 16);
#pragma unroll
            for (int nt = 0; nt < 4; ++nt) {
                const v16h B = *(const v16h*)&sW2[((nt * 2 + c) * 32 + lane) * 16];
                acc2[nt] = __builtin_amdgcn_wmma_f32_16x16x32_f16(
                    false, A.v, false, B, (short)0, acc2[nt], false, false);
            }
        }

        // ---------------- layer 3: [16x64] x [64x1] + epilogue -----------
        float part[8];
#pragma unroll
        for (int r = 0; r < 8; ++r) part[r] = 0.0f;
#pragma unroll
        for (int nt = 0; nt < 4; ++nt) {
#pragma unroll
            for (int r = 0; r < 8; ++r) {
                float v = acc2[nt][r] + rb2[nt];
                v = v > 0.0f ? v : 0.0f;
                part[r] += v * rw3[nt];
            }
        }
        // reduce across the 16-lane half-groups (masks 1,2,4,8 stay in-group)
#pragma unroll
        for (int s = 1; s <= 8; s <<= 1) {
#pragma unroll
            for (int r = 0; r < 8; ++r)
                part[r] += __shfl_xor(part[r], s, 32);
        }

        // Redistribute u values with full EXEC (bpermute reads need active
        // source lanes): lane group hi needs u from lanes hi*8 + r.
        float uvr[8];
#pragma unroll
        for (int r = 0; r < 8; ++r)
            uvr[r] = __shfl(uval, hi * 8 + r, 32);

        if (m == 0) {   // lane 0 -> edges t*16+0..7 ; lane 16 -> +8..15
            const int ebase = t * 16 + hi * 8;
#pragma unroll
            for (int r = 0; r < 8; ++r) {
                const int e2 = ebase + r;
                if (e2 < E) {
                    float g = -logf(-logf(uvr[r] + EPS_G) + EPS_G);
                    float x = (part[r] + bias3 + g) * TEMP_INV;
                    out[e2] = 1.0f / (1.0f + expf(-x));
                }
            }
        }

        nrow = nrow_n;
        ncol = ncol_n;
    }
}

// ---------------------------------------------------------------------------
extern "C" void kernel_launch(void* const* d_in, const int* in_sizes, int n_in,
                              void* d_out, int out_size, void* d_ws, size_t ws_size,
                              hipStream_t stream) {
    const float* emb = (const float*)d_in[0];
    const int*   ei  = (const int*)d_in[1];
    const float* u   = (const float*)d_in[2];
    const float* W1  = (const float*)d_in[3];
    const float* b1  = (const float*)d_in[4];
    const float* W2  = (const float*)d_in[5];
    const float* b2  = (const float*)d_in[6];
    const float* W3  = (const float*)d_in[7];
    const float* b3  = (const float*)d_in[8];
    float* out = (float*)d_out;

    const long long NF = in_sizes[0];      // N * 128
    const int E = in_sizes[2];

    char* ws = (char*)d_ws;
    _Float16* embh = (_Float16*)ws;
    size_t off = ((size_t)NF * 2 + 255) & ~(size_t)255;
    _Float16* w1p = (_Float16*)(ws + off);
    off += (size_t)(4 * 8 * 32 * 16) * 2;  // 32 KB
    _Float16* w2p = (_Float16*)(ws + off);

    pgx_cvt_f16<<<2048, 256, 0, stream>>>(emb, embh, NF);
    pgx_pack_w<<<(4 * 8 * 32 * 16 + 255) / 256, 256, 0, stream>>>(W1, w1p, 8);
    pgx_pack_w<<<(4 * 2 * 32 * 16 + 255) / 256, 256, 0, stream>>>(W2, w2p, 2);

    const int numTiles = (E + 15) / 16;
    int blocks = (numTiles + 7) / 8;
    if (blocks > 4096) blocks = 4096;
    if (blocks < 1) blocks = 1;
    pgx_main<<<blocks, 256, 0, stream>>>(embh, ei, u, w1p, w2p,
                                         b1, b2, W3, b3, out, E, numTiles);
}
